// HAMSnet_85117661872780
// MI455X (gfx1250) — matrix operand order repaired and checked by reading, not compiled, not run on hardware
//
#include <hip/hip_runtime.h>
#include <hip/hip_bf16.h>
#include <math.h>

#define BB 512
#define CC 64
#define LL 640

typedef float    v2f  __attribute__((ext_vector_type(2)));
typedef float    v8f  __attribute__((ext_vector_type(8)));
typedef _Float16 v16h __attribute__((ext_vector_type(16)));

#if defined(__has_builtin)
#if __has_builtin(__builtin_amdgcn_wmma_f32_16x16x4_f32)
#define HAS_WMMA_F32X4 1
#endif
#endif
#ifndef HAS_WMMA_F32X4
#define HAS_WMMA_F32X4 0
#endif

__device__ __forceinline__ float frelu(float v) { return v > 0.f ? v : 0.f; }

// ---------------------------------------------------------------------------
// conv1d: y[b,co,l] = bias[co] + sum_{ci,t} w[co,ci,t]*x[b,ci,2*l-pad+t]
// One block = one (b, 64-output chunk). Input tile staged in LDS.
// ---------------------------------------------------------------------------
__global__ void k_conv1d(const float* __restrict__ x, const float* __restrict__ w,
                         const float* __restrict__ bias, float* __restrict__ y,
                         int Lin, int Lout, int Kw, int pad) {
    int nch = (Lout + 63) >> 6;
    int b = blockIdx.x / nch;
    int chunk = blockIdx.x % nch;
    int l0 = chunk * 64;
    int span = 2 * 64 + Kw - 1;              // <= 130
    __shared__ float xs[64 * 132];
    const float* xb = x + (size_t)b * CC * Lin;
    int total = 64 * span;
    for (int i = threadIdx.x; i < total; i += 256) {
        int ch = i / span, j = i - ch * span;
        int pos = 2 * l0 - pad + j;
        xs[ch * 132 + j] = (pos >= 0 && pos < Lin) ? xb[(size_t)ch * Lin + pos] : 0.f;
    }
    __syncthreads();
    for (int i = threadIdx.x; i < 4096; i += 256) {
        int co = i >> 6, l = i & 63;
        int lo = l0 + l;
        if (lo >= Lout) continue;
        float acc = bias[co];
        const float* wr = w + (size_t)co * CC * Kw;
        for (int ci = 0; ci < CC; ++ci) {
            int base = ci * 132 + 2 * l;
            for (int t = 0; t < Kw; ++t) acc += wr[ci * Kw + t] * xs[base + t];
        }
        y[((size_t)b * CC + co) * Lout + lo] = acc;
    }
}

// ---------------------------------------------------------------------------
// enc = maxpool8(x) + maxpool4(enc1) + maxpool2(enc2) + enc3   -> [B,64,80]
// ---------------------------------------------------------------------------
__global__ void k_poolsum(const float* __restrict__ x, const float* __restrict__ e1,
                          const float* __restrict__ e2, const float* __restrict__ e3,
                          float* __restrict__ enc) {
    long t = (long)blockIdx.x * blockDim.x + threadIdx.x;
    long total = (long)BB * CC * 80;
    if (t >= total) return;
    int i = (int)(t % 80);
    long bc = t / 80;
    const float* p8 = x + bc * LL + i * 8;
    float m8 = p8[0];
#pragma unroll
    for (int j = 1; j < 8; ++j) m8 = fmaxf(m8, p8[j]);
    const float* p4 = e1 + bc * 320 + i * 4;
    float m4 = p4[0];
#pragma unroll
    for (int j = 1; j < 4; ++j) m4 = fmaxf(m4, p4[j]);
    const float* p2 = e2 + bc * 160 + i * 2;
    float m2 = fmaxf(p2[0], p2[1]);
    enc[t] = m8 + m4 + m2 + e3[bc * 80 + i];
}

// ---------------------------------------------------------------------------
// feat1/feat2: depthwise k3 (pad1; only middle 3x3 row touches H=1 data) over
// 64 ch, then 64->4 pointwise + relu, twice (dw5a/pw5a, dw5c/pw5c).
// Block = (b, 128-position chunk); input slab staged in LDS (64 x 130).
// ---------------------------------------------------------------------------
__global__ void k_feat12(const float* __restrict__ f,
                         const float* __restrict__ dwa, const float* __restrict__ pwa,
                         const float* __restrict__ dwc, const float* __restrict__ pwc,
                         float* __restrict__ f1, float* __restrict__ f2, int N) {
    int nch = (N + 127) >> 7;
    int b = blockIdx.x / nch;
    int c0 = (blockIdx.x % nch) << 7;
    int cols = N - c0; if (cols > 128) cols = 128;
    __shared__ float xs[64 * 132];           // positions c0-1 .. c0+128
    const float* xb = f + (size_t)b * CC * N;
    int span = 130;
    for (int i = threadIdx.x; i < 64 * span; i += 128) {
        int ch = i / span, j = i - ch * span;
        int pos = c0 - 1 + j;
        xs[ch * 132 + j] = (pos >= 0 && pos < N) ? xb[(size_t)ch * N + pos] : 0.f;
    }
    __syncthreads();
    int tid = threadIdx.x;
    if (tid >= cols) return;
    int n = c0 + tid;
    float acc1[4] = {0, 0, 0, 0}, acc2[4] = {0, 0, 0, 0};
    for (int ch = 0; ch < CC; ++ch) {
        float xm = xs[ch * 132 + tid];
        float x0 = xs[ch * 132 + tid + 1];
        float xp = xs[ch * 132 + tid + 2];
        float ta = dwa[ch * 9 + 3] * xm + dwa[ch * 9 + 4] * x0 + dwa[ch * 9 + 5] * xp;
        float tc = dwc[ch * 9 + 3] * xm + dwc[ch * 9 + 4] * x0 + dwc[ch * 9 + 5] * xp;
#pragma unroll
        for (int c = 0; c < 4; ++c) {
            acc1[c] += pwa[c * CC + ch] * ta;
            acc2[c] += pwc[c * CC + ch] * tc;
        }
    }
#pragma unroll
    for (int c = 0; c < 4; ++c) {
        f1[((size_t)b * 4 + c) * N + n] = frelu(acc1[c]);
        f2[((size_t)b * 4 + c) * N + n] = frelu(acc2[c]);
    }
}

// ---------------------------------------------------------------------------
// q/k/v projections: per-channel scale (depthwise 1x1) + 4x4 pointwise.
// ---------------------------------------------------------------------------
__global__ void k_qkv(const float* __restrict__ f1,
                      const float* __restrict__ qdw, const float* __restrict__ qpw,
                      const float* __restrict__ kdw, const float* __restrict__ kpw,
                      const float* __restrict__ vdw, const float* __restrict__ vpw,
                      float* __restrict__ q, float* __restrict__ k, float* __restrict__ v,
                      int N) {
    long t = (long)blockIdx.x * blockDim.x + threadIdx.x;
    if (t >= (long)BB * N) return;
    int b = (int)(t / N), n = (int)(t % N);
    float fv[4];
#pragma unroll
    for (int c = 0; c < 4; ++c) fv[c] = f1[((size_t)b * 4 + c) * N + n];
#pragma unroll
    for (int c = 0; c < 4; ++c) {
        float aq = 0.f, ak = 0.f, av = 0.f;
#pragma unroll
        for (int d = 0; d < 4; ++d) {
            aq += qpw[c * 4 + d] * qdw[d] * fv[d];
            ak += kpw[c * 4 + d] * kdw[d] * fv[d];
            av += vpw[c * 4 + d] * vdw[d] * fv[d];
        }
        size_t o = ((size_t)b * 4 + c) * N + n;
        q[o] = aq; k[o] = ak; v[o] = av;
    }
}

// ---------------------------------------------------------------------------
// PAM: attention over [N,N] with inner dim 4, two-pass softmax.
// One wave per (batch, 16-row query block). Energy tile = WMMA F32 16x16x4.
// Pass 1: per-lane running max over all m-tiles (VALU only), one cross-lane
//         reduction at the end. Pass 2: recompute tiles (WMMA is cheap at
//         K=4), exp with uniform row max, per-lane accumulation; one final
//         cross-lane reduction. No DS traffic inside the m-loops.
// D layout (8 VGPRs): lane l, vgpr r -> E[n = r + 8*(l>>4)][m = l&15].
// out = gpam * softmax(E) @ v^T + feat1
// ---------------------------------------------------------------------------
__global__ void k_pam(const float* __restrict__ q, const float* __restrict__ k,
                      const float* __restrict__ v, const float* __restrict__ f1,
                      const float* __restrict__ gpam, float* __restrict__ out, int N) {
    int wpb = blockDim.x >> 5;
    int wid = blockIdx.x * wpb + (threadIdx.x >> 5);
    int lane = threadIdx.x & 31;
    int nb = N >> 4;
    if (wid >= BB * nb) return;                 // uniform per wave: EXEC stays full
    int b = wid / nb;
    int nbase = (wid % nb) << 4;
    int half = lane >> 4, l16 = lane & 15;

    const float* qb = q + (size_t)b * 4 * N;
    const float* kb = k + (size_t)b * 4 * N;
    const float* vb = v + (size_t)b * 4 * N;

#if HAS_WMMA_F32X4
    // A (16x4 f32): lanes 0-15 hold K=0,1 ; lanes 16-31 hold K=2,3 (ISA 7.12.2)
    v2f a;
    a.x = qb[(size_t)(2 * half + 0) * N + nbase + l16];
    a.y = qb[(size_t)(2 * half + 1) * N + nbase + l16];
#else
    float qrow[4][8];
#pragma unroll
    for (int c = 0; c < 4; ++c)
#pragma unroll
        for (int r = 0; r < 8; ++r)
            qrow[c][r] = qb[(size_t)c * N + nbase + r + 8 * half];
#endif

    // ---- pass 1: row maxima ----
    float rmax[8];
#pragma unroll
    for (int r = 0; r < 8; ++r) rmax[r] = -1e30f;
    for (int mb = 0; mb < nb; ++mb) {
        int mbase = mb << 4;
#if HAS_WMMA_F32X4
        v2f bm;
        bm.x = kb[(size_t)(half + 0) * N + mbase + l16];
        bm.y = kb[(size_t)(half + 2) * N + mbase + l16];
        v8f cz = {0.f, 0.f, 0.f, 0.f, 0.f, 0.f, 0.f, 0.f};
        v8f e = __builtin_amdgcn_wmma_f32_16x16x4_f32(false, a, false, bm,
                                                      (short)0, cz, false, false);
#pragma unroll
        for (int r = 0; r < 8; ++r) rmax[r] = fmaxf(rmax[r], e[r]);
#else
        float kc[4];
#pragma unroll
        for (int c = 0; c < 4; ++c) kc[c] = kb[(size_t)c * N + mbase + l16];
#pragma unroll
        for (int r = 0; r < 8; ++r)
            rmax[r] = fmaxf(rmax[r], qrow[0][r] * kc[0] + qrow[1][r] * kc[1] +
                                     qrow[2][r] * kc[2] + qrow[3][r] * kc[3]);
#endif
    }
#pragma unroll
    for (int r = 0; r < 8; ++r) {              // one reduction over 16-lane half
        rmax[r] = fmaxf(rmax[r], __shfl_xor(rmax[r], 1, 32));
        rmax[r] = fmaxf(rmax[r], __shfl_xor(rmax[r], 2, 32));
        rmax[r] = fmaxf(rmax[r], __shfl_xor(rmax[r], 4, 32));
        rmax[r] = fmaxf(rmax[r], __shfl_xor(rmax[r], 8, 32));
    }

    // ---- pass 2: exp-accumulate with uniform row max ----
    float rsum[8], oacc[4][8];
#pragma unroll
    for (int r = 0; r < 8; ++r) {
        rsum[r] = 0.f;
#pragma unroll
        for (int c = 0; c < 4; ++c) oacc[c][r] = 0.f;
    }
    for (int mb = 0; mb < nb; ++mb) {
        int mbase = mb << 4;
        if (mb + 1 < nb)
            __builtin_prefetch(kb + (size_t)half * N + mbase + 16, 0, 1);
        float ev[8];
#if HAS_WMMA_F32X4
        v2f bm;
        bm.x = kb[(size_t)(half + 0) * N + mbase + l16];
        bm.y = kb[(size_t)(half + 2) * N + mbase + l16];
        v8f cz = {0.f, 0.f, 0.f, 0.f, 0.f, 0.f, 0.f, 0.f};
        v8f e = __builtin_amdgcn_wmma_f32_16x16x4_f32(false, a, false, bm,
                                                      (short)0, cz, false, false);
#pragma unroll
        for (int r = 0; r < 8; ++r) ev[r] = e[r];
#else
        float kc[4];
#pragma unroll
        for (int c = 0; c < 4; ++c) kc[c] = kb[(size_t)c * N + mbase + l16];
#pragma unroll
        for (int r = 0; r < 8; ++r)
            ev[r] = qrow[0][r] * kc[0] + qrow[1][r] * kc[1] +
                    qrow[2][r] * kc[2] + qrow[3][r] * kc[3];
#endif
        float vv[4];
#pragma unroll
        for (int c = 0; c < 4; ++c) vv[c] = vb[(size_t)c * N + mbase + l16];
#pragma unroll
        for (int r = 0; r < 8; ++r) {
            float p = __expf(ev[r] - rmax[r]);
            rsum[r] += p;
#pragma unroll
            for (int c = 0; c < 4; ++c) oacc[c][r] += p * vv[c];
        }
    }
    // final cross-lane reductions over the 16-lane half (m direction)
#pragma unroll
    for (int r = 0; r < 8; ++r) {
        for (int m = 1; m < 16; m <<= 1) rsum[r] += __shfl_xor(rsum[r], m, 32);
#pragma unroll
        for (int c = 0; c < 4; ++c)
            for (int m = 1; m < 16; m <<= 1) oacc[c][r] += __shfl_xor(oacc[c][r], m, 32);
    }
    float g = gpam[0];
    if (l16 < 8) {
        int r = l16;
        int n = nbase + r + 8 * half;
        const float* fb = f1 + (size_t)b * 4 * N;
        float* ob = out + (size_t)b * 4 * N;
#pragma unroll
        for (int c = 0; c < 4; ++c) {
            float o = oacc[c][r] / rsum[r];
            ob[(size_t)c * N + n] = g * o + fb[(size_t)c * N + n];
        }
    }
}

// ---------------------------------------------------------------------------
// 4-channel depthwise k3 (pad1) + 4x4 pointwise + relu  (dw51/pw51, dw52/pw52)
// ---------------------------------------------------------------------------
__global__ void k_dwpw4(const float* __restrict__ in, const float* __restrict__ dw,
                        const float* __restrict__ pw, float* __restrict__ out, int N) {
    long t = (long)blockIdx.x * blockDim.x + threadIdx.x;
    if (t >= (long)BB * N) return;
    int b = (int)(t / N), n = (int)(t % N);
    const float* ib = in + (size_t)b * 4 * N;
    float tt[4];
#pragma unroll
    for (int c = 0; c < 4; ++c) {
        const float* r = ib + (size_t)c * N;
        float xm = (n > 0) ? r[n - 1] : 0.f;
        float x0 = r[n];
        float xp = (n < N - 1) ? r[n + 1] : 0.f;
        tt[c] = dw[c * 9 + 3] * xm + dw[c * 9 + 4] * x0 + dw[c * 9 + 5] * xp;
    }
#pragma unroll
    for (int c2 = 0; c2 < 4; ++c2) {
        float acc = 0.f;
#pragma unroll
        for (int c = 0; c < 4; ++c) acc += pw[c2 * 4 + c] * tt[c];
        out[((size_t)b * 4 + c2) * N + n] = frelu(acc);
    }
}

// ---------------------------------------------------------------------------
// CAM: 4x4 Gram, energy_new = rowmax - energy, softmax, out = attn@f + resid
// One block per batch.
// ---------------------------------------------------------------------------
__global__ void k_cam(const float* __restrict__ f, const float* __restrict__ gcam,
                      float* __restrict__ out, int N) {
    int b = blockIdx.x, tid = threadIdx.x;         // blockDim = 128
    __shared__ float red[128 * 16];
    __shared__ float attn[16];
    const float* fb = f + (size_t)b * 4 * N;
    float g[16];
#pragma unroll
    for (int j = 0; j < 16; ++j) g[j] = 0.f;
    for (int n = tid; n < N; n += 128) {
        float fv[4];
#pragma unroll
        for (int c = 0; c < 4; ++c) fv[c] = fb[(size_t)c * N + n];
#pragma unroll
        for (int c = 0; c < 4; ++c)
#pragma unroll
            for (int d = 0; d < 4; ++d) g[c * 4 + d] += fv[c] * fv[d];
    }
#pragma unroll
    for (int j = 0; j < 16; ++j) red[j * 128 + tid] = g[j];
    __syncthreads();
    if (tid < 16) {
        float s = 0.f;
        for (int i = 0; i < 128; ++i) s += red[tid * 128 + i];
        red[tid] = s;                               // energy[c*4+d]
    }
    __syncthreads();
    if (tid < 4) {
        int c = tid;
        float m = -1e30f;
#pragma unroll
        for (int d = 0; d < 4; ++d) m = fmaxf(m, red[c * 4 + d]);
        float mx = -1e30f;
#pragma unroll
        for (int d = 0; d < 4; ++d) mx = fmaxf(mx, m - red[c * 4 + d]);
        float ex[4], s = 0.f;
#pragma unroll
        for (int d = 0; d < 4; ++d) { ex[d] = __expf((m - red[c * 4 + d]) - mx); s += ex[d]; }
#pragma unroll
        for (int d = 0; d < 4; ++d) attn[c * 4 + d] = ex[d] / s;
    }
    __syncthreads();
    float gc = gcam[0];
    float at[16];
#pragma unroll
    for (int j = 0; j < 16; ++j) at[j] = attn[j];
    float* ob = out + (size_t)b * 4 * N;
    for (int n = tid; n < N; n += 128) {
        float fv[4];
#pragma unroll
        for (int c = 0; c < 4; ++c) fv[c] = fb[(size_t)c * N + n];
#pragma unroll
        for (int c = 0; c < 4; ++c) {
            float o = 0.f;
#pragma unroll
            for (int d = 0; d < 4; ++d) o += at[c * 4 + d] * fv[d];
            ob[(size_t)c * N + n] = gc * o + fv[c];
        }
    }
}

// ---------------------------------------------------------------------------
// Fused ADAF tail: u = relu(pw6·(aws0∘dw6∘sa)) + relu(pw7·(aws1∘dw7∘sc));
// X = dw8∘u; Y = relu(pw8·X) [+ resid].  pw8 GEMM = WMMA F32 16x16x32 F16.
// Block: 128 threads (4 waves), 64 output columns; wave mt owns rows 16mt..+15.
// ---------------------------------------------------------------------------
__global__ void k_fuse(const float* __restrict__ sa, const float* __restrict__ sc,
                       const float* __restrict__ resid,
                       const float* __restrict__ pw6, const float* __restrict__ pw7,
                       const float* __restrict__ dw6, const float* __restrict__ dw7,
                       const float* __restrict__ dw8, const float* __restrict__ pw8,
                       const float* __restrict__ aw, float* __restrict__ dst, int N) {
    __shared__ _Float16 Wh[64 * 64];        // [m][k]
    __shared__ _Float16 Xh[64 * 72];        // [k][col], stride 72
    __shared__ float s_sa[4][64], s_sc[4][64];
    int tid = threadIdx.x;
    long tile0 = (long)blockIdx.x * 64;
    for (int i = tid; i < 4096; i += 128) Wh[i] = (_Float16)pw8[i];
    float aws0 = 1.f / (1.f + __expf(-aw[0]));
    float aws1 = 1.f / (1.f + __expf(-aw[1]));
    if (tid < 64) {
        long t = tile0 + tid;
        int b = (int)(t / N), n = (int)(t % N);
        const float* sab = sa + (size_t)b * 4 * N + n;
        const float* scb = sc + (size_t)b * 4 * N + n;
#pragma unroll
        for (int c = 0; c < 4; ++c) {
            s_sa[c][tid] = sab[(size_t)c * N];
            s_sc[c][tid] = scb[(size_t)c * N];
        }
    }
    __syncthreads();
    for (int i = tid; i < 4096; i += 128) {
        int ch = i >> 6, col = i & 63;
        float u = 0.f, w = 0.f;
#pragma unroll
        for (int c = 0; c < 4; ++c) {
            u += pw6[ch * 4 + c] * dw6[c] * aws0 * s_sa[c][col];
            w += pw7[ch * 4 + c] * dw7[c] * aws1 * s_sc[c][col];
        }
        Xh[ch * 72 + col] = (_Float16)(dw8[ch] * (frelu(u) + frelu(w)));
    }
    __syncthreads();
    int wave = tid >> 5, lane = tid & 31, half = lane >> 4, l16 = lane & 15;
    int mt = wave;
    // A tiles: f16 16x32 layout (ISA 7.12.2): pair-packed K per VGPR
    v16h A[2];
#pragma unroll
    for (int ks = 0; ks < 2; ++ks) {
        int m = mt * 16 + l16;
#pragma unroll
        for (int vp = 0; vp < 8; ++vp) {
            int kk = (vp < 4) ? (ks * 32 + half * 8 + 2 * vp)
                              : (ks * 32 + 16 + half * 8 + 2 * (vp - 4));
            A[ks][2 * vp]     = Wh[m * 64 + kk];
            A[ks][2 * vp + 1] = Wh[m * 64 + kk + 1];
        }
    }
#pragma unroll
    for (int ct = 0; ct < 4; ++ct) {
        v16h Bt[2];
#pragma unroll
        for (int ks = 0; ks < 2; ++ks) {
#pragma unroll
            for (int h = 0; h < 16; ++h) {
                int kk = (h < 8) ? (ks * 32 + half * 8 + h)
                                 : (ks * 32 + 16 + half * 8 + (h - 8));
                Bt[ks][h] = Xh[kk * 72 + ct * 16 + l16];
            }
        }
        v8f acc = {0.f, 0.f, 0.f, 0.f, 0.f, 0.f, 0.f, 0.f};
        acc = __builtin_amdgcn_wmma_f32_16x16x32_f16(false, A[0], false, Bt[0],
                                                     (short)0, acc, false, false);
        acc = __builtin_amdgcn_wmma_f32_16x16x32_f16(false, A[1], false, Bt[1],
                                                     (short)0, acc, false, false);
        long t = tile0 + ct * 16 + l16;
        int b = (int)(t / N), n = (int)(t % N);
        float* db = dst + (size_t)b * CC * N + n;
        const float* rb = resid ? (resid + (size_t)b * CC * N + n) : nullptr;
#pragma unroll
        for (int r = 0; r < 8; ++r) {
            int m = mt * 16 + r + 8 * half;
            float y = frelu(acc[r]);
            if (rb) y += rb[(size_t)m * N];
            db[(size_t)m * N] = y;
        }
    }
}

// ---------------------------------------------------------------------------
// dst[b,c,i] = src[b,c,(i*Ls)/Ld] (+ add1[b,c,i]) (+ add2[b,c,i])
// Ls == Ld gives a plain (aliased-safe) elementwise add.
// ---------------------------------------------------------------------------
__global__ void k_gather_add(const float* __restrict__ src, int Ls,
                             const float* __restrict__ add1,
                             const float* __restrict__ add2,
                             float* __restrict__ dst, int Ld) {
    long t = (long)blockIdx.x * blockDim.x + threadIdx.x;
    long total = (long)BB * CC * Ld;
    if (t >= total) return;
    int i = (int)(t % Ld);
    long bc = t / Ld;
    float vv = src[bc * Ls + (int)(((long)i * Ls) / Ld)];
    if (add1) vv += add1[t];
    if (add2) vv += add2[t];
    dst[t] = vv;
}

// ---------------------------------------------------------------------------
extern "C" void kernel_launch(void* const* d_in, const int* in_sizes, int n_in,
                              void* d_out, int out_size, void* d_ws, size_t ws_size,
                              hipStream_t stream) {
    const float* x    = (const float*)d_in[0];
    const float* w1   = (const float*)d_in[1];
    const float* b1   = (const float*)d_in[2];
    const float* w2   = (const float*)d_in[3];
    const float* b2   = (const float*)d_in[4];
    const float* w3   = (const float*)d_in[5];
    const float* b3   = (const float*)d_in[6];
    const float* dw5a = (const float*)d_in[7];
    const float* pw5a = (const float*)d_in[8];
    const float* dw5c = (const float*)d_in[9];
    const float* pw5c = (const float*)d_in[10];
    const float* qdw  = (const float*)d_in[11];
    const float* qpw  = (const float*)d_in[12];
    const float* kdw  = (const float*)d_in[13];
    const float* kpw  = (const float*)d_in[14];
    const float* vdw  = (const float*)d_in[15];
    const float* vpw  = (const float*)d_in[16];
    const float* gpam = (const float*)d_in[17];
    const float* gcam = (const float*)d_in[18];
    const float* dw51 = (const float*)d_in[19];
    const float* pw51 = (const float*)d_in[20];
    const float* dw52 = (const float*)d_in[21];
    const float* pw52 = (const float*)d_in[22];
    const float* dw6  = (const float*)d_in[23];
    const float* pw6  = (const float*)d_in[24];
    const float* dw7  = (const float*)d_in[25];
    const float* pw7  = (const float*)d_in[26];
    const float* dw8  = (const float*)d_in[27];
    const float* pw8  = (const float*)d_in[28];
    const float* aw   = (const float*)d_in[29];
    float* out = (float*)d_out;

    // workspace arena (fp32 bump allocator)
    float* wsf = (float*)d_ws;
    size_t off = 0;
    auto alloc = [&](size_t n) { float* p = wsf + off; off += n; return p; };
    float* E1  = alloc((size_t)BB * CC * 320);
    float* E2  = alloc((size_t)BB * CC * 160);
    float* E3  = alloc((size_t)BB * CC * 80);
    float* ENC = alloc((size_t)BB * CC * 80);
    float* G1  = alloc((size_t)BB * CC * 80);
    float* G2  = alloc((size_t)BB * CC * 80);
    float* D00 = alloc((size_t)BB * CC * 80);
    float* D11 = alloc((size_t)BB * CC * 160);
    float* D22 = alloc((size_t)BB * CC * 320);
    float* D33 = alloc((size_t)BB * CC * 640);
    float* F1  = alloc((size_t)BB * 4 * 640);
    float* F2  = alloc((size_t)BB * 4 * 640);
    float* Qb  = alloc((size_t)BB * 4 * 640);
    float* Kb  = alloc((size_t)BB * 4 * 640);
    float* Vb  = alloc((size_t)BB * 4 * 640);
    float* PO  = alloc((size_t)BB * 4 * 640);
    float* CO  = alloc((size_t)BB * 4 * 640);
    float* SA  = alloc((size_t)BB * 4 * 640);
    float* SC  = alloc((size_t)BB * 4 * 640);
    if (off * sizeof(float) > ws_size) return;   // insufficient scratch

    auto gridE = [](long n) { return (unsigned)((n + 255) / 256); };

    // one full ADAF block: dst = [resid +] adaf(src)
    auto adaf = [&](const float* src, float* dst, const float* resid, int N) {
        long T = (long)BB * N;
        unsigned fblocks = (unsigned)(BB * ((N + 127) >> 7));
        k_feat12<<<fblocks, 128, 0, stream>>>(src, dw5a, pw5a, dw5c, pw5c, F1, F2, N);
        k_qkv<<<gridE(T), 256, 0, stream>>>(F1, qdw, qpw, kdw, kpw, vdw, vpw, Qb, Kb, Vb, N);
        int waves = BB * (N >> 4);
        k_pam<<<(waves + 7) / 8, 256, 0, stream>>>(Qb, Kb, Vb, F1, gpam, PO, N);
        k_dwpw4<<<gridE(T), 256, 0, stream>>>(PO, dw51, pw51, SA, N);
        k_cam<<<BB, 128, 0, stream>>>(F2, gcam, CO, N);
        k_dwpw4<<<gridE(T), 256, 0, stream>>>(CO, dw52, pw52, SC, N);
        unsigned tiles = (unsigned)(T / 64);     // B*N always a multiple of 64
        k_fuse<<<tiles, 128, 0, stream>>>(SA, SC, resid, pw6, pw7, dw6, dw7, dw8,
                                          pw8, aw, dst, N);
    };

    // ---- encoder ----
    k_conv1d<<<BB * 5, 256, 0, stream>>>(x,  w1, b1, E1, 640, 320, 3, 1);
    k_conv1d<<<BB * 3, 256, 0, stream>>>(E1, w2, b2, E2, 320, 160, 2, 0);
    k_conv1d<<<BB * 2, 256, 0, stream>>>(E2, w3, b3, E3, 160,  80, 2, 0);
    k_poolsum<<<gridE((long)BB * CC * 80), 256, 0, stream>>>(x, E1, E2, E3, ENC);

    // ---- block1 x3 (pure), then enc = enc + e ----
    adaf(ENC, G1, nullptr, 80);
    adaf(G1,  G2, nullptr, 80);
    adaf(G2,  G1, nullptr, 80);
    k_gather_add<<<gridE((long)BB * CC * 80), 256, 0, stream>>>(ENC, 80, G1, nullptr, G2, 80);

    // ---- dec0 = enc + adaf(enc) (in place in G2) ----
    adaf(G2, G2, G2, 80);

    // ---- decoder splits ----
    k_gather_add<<<gridE((long)BB * CC * 80),  256, 0, stream>>>(G2, 80, E3, nullptr, D00, 80);
    k_gather_add<<<gridE((long)BB * CC * 160), 256, 0, stream>>>(G2, 80, E2, nullptr, D11, 160);
    k_gather_add<<<gridE((long)BB * CC * 320), 256, 0, stream>>>(G2, 80, E1, nullptr, D22, 320);
    k_gather_add<<<gridE((long)BB * CC * 640), 256, 0, stream>>>(G2, 80, x,  nullptr, D33, 640);

    adaf(D00, D00, D00, 80);
    adaf(D11, D11, D11, 160);
    adaf(D22, D22, D22, 320);
    adaf(D33, D33, D33, 640);

    // ---- decoder fusion + final output ----
    k_gather_add<<<gridE((long)BB * CC * 160), 256, 0, stream>>>(D00, 80,  D11, nullptr, D11, 160);
    k_gather_add<<<gridE((long)BB * CC * 320), 256, 0, stream>>>(D11, 160, D22, nullptr, D22, 320);
    k_gather_add<<<gridE((long)BB * CC * 640), 256, 0, stream>>>(D22, 320, D33, x, out, 640);
}
